// Encoder_506806141403
// MI455X (gfx1250) — compile-verified, matrix operand order checked by reading, not run on hardware
//
#include <hip/hip_runtime.h>
#include <cmath>

// ---------------- problem constants ----------------
#define DIM    768
#define NHEAD  12
#define NLAYER 12
#define FFDIM  3072
#define SEQ    1024
#define BATCH  4
#define MTOK   (BATCH*SEQ)     // 4096 rows of activations
#define HD     64              // head dim
#define ATT_SCALE 0.03608439182435161f   // 768^-0.5 (reference scales by embed_dim^-0.5)

typedef __attribute__((ext_vector_type(16))) _Float16 v16h;
typedef __attribute__((ext_vector_type(8)))  float    v8f;

// ---------------- WMMA helpers (CDNA5 gfx1250, wave32) ----------------
__device__ __forceinline__ v8f wmma16(v16h a, v16h b, v8f c) {
  // D(16x16,f32) = A(16x32,f16) * B(32x16,f16) + C
  return __builtin_amdgcn_wmma_f32_16x16x32_f16(false, a, false, b, (short)0, c, false, false);
}

// K index for A/B fragment element pair i (ISA 7.12.2, 16-bit 16x32 A layout):
// lanes 0-15 hold K {0..7,16..23}; lanes 16-31 hold K {8..15,24..31}, 2 halves/VGPR.
__device__ __forceinline__ int frag_k0(int i, int g) {
  return (i < 4) ? (g * 8 + 2 * i) : (16 + g * 8 + 2 * (i - 4));
}

// A fragment from row-major f16 [16 x >=32] at p with row stride (halves)
__device__ __forceinline__ v16h frag_a(const _Float16* p, int stride, int lane) {
  int m = lane & 15, g = lane >> 4;
  v16h a;
#pragma unroll
  for (int i = 0; i < 8; ++i) {
    int k0 = frag_k0(i, g);
    a[2 * i]     = p[m * stride + k0];
    a[2 * i + 1] = p[m * stride + k0 + 1];
  }
  return a;
}

// B fragment: memory holds B[k][n] row-major (stride = row stride in halves)
__device__ __forceinline__ v16h frag_b(const _Float16* p, int stride, int lane) {
  int n = lane & 15, g = lane >> 4;
  v16h b;
#pragma unroll
  for (int i = 0; i < 8; ++i) {
    int k0 = frag_k0(i, g);
    b[2 * i]     = p[k0 * stride + n];
    b[2 * i + 1] = p[(k0 + 1) * stride + n];
  }
  return b;
}

// B fragment where memory holds the transpose: mem[n][k] (stride over n rows)
__device__ __forceinline__ v16h frag_bt(const _Float16* p, int stride, int lane) {
  int n = lane & 15, g = lane >> 4;
  v16h b;
#pragma unroll
  for (int i = 0; i < 8; ++i) {
    int k0 = frag_k0(i, g);
    b[2 * i]     = p[n * stride + k0];
    b[2 * i + 1] = p[n * stride + k0 + 1];
  }
  return b;
}

// ---------------- LayerNorm: fp32 in -> f16 out ----------------
__global__ __launch_bounds__(256)
void ln_kernel(const float* __restrict__ x, const float* __restrict__ gg,
               const float* __restrict__ bb, _Float16* __restrict__ out) {
  const int row = blockIdx.x;
  const float* xr = x + (size_t)row * DIM;
  const int tid = threadIdx.x;
  float s = 0.f, s2 = 0.f;
  for (int j = tid; j < DIM; j += 256) { float v = xr[j]; s += v; s2 += v * v; }
#pragma unroll
  for (int m = 16; m; m >>= 1) { s += __shfl_xor(s, m); s2 += __shfl_xor(s2, m); }
  __shared__ float sh[16];
  const int w = tid >> 5, lane = tid & 31;
  if (lane == 0) { sh[w] = s; sh[8 + w] = s2; }
  __syncthreads();
  if (w == 0) {
    float a = (lane < 8) ? sh[lane] : 0.f;
    float c = (lane < 8) ? sh[8 + lane] : 0.f;
#pragma unroll
    for (int m = 4; m; m >>= 1) { a += __shfl_xor(a, m); c += __shfl_xor(c, m); }
    if (lane == 0) { sh[0] = a; sh[1] = c; }
  }
  __syncthreads();
  const float mu = sh[0] * (1.f / DIM);
  const float var = sh[1] * (1.f / DIM) - mu * mu;
  const float rs = rsqrtf(var + 1e-5f);
  for (int j = tid; j < DIM; j += 256)
    out[(size_t)row * DIM + j] = (_Float16)(((xr[j] - mu) * rs) * gg[j] + bb[j]);
}

// ---------------- tile staging for GEMM (vectorized, double-buffer target) ----
// A tile: 128x32 f16 ; W tile: 32x128 f32->f16
__device__ __forceinline__ void stage_tiles(const _Float16* __restrict__ A,
                                            const float* __restrict__ W,
                                            int K, int N, int row0, int col0, int k0,
                                            int tid,
                                            _Float16 (*As)[40], _Float16 (*Ws)[136]) {
#pragma unroll
  for (int e = tid; e < 1024; e += 256) {          // 128 rows x 8 uint2 (4 halves)
    int rr = e >> 3, cc = e & 7;
    uint2 d = *reinterpret_cast<const uint2*>(A + (size_t)(row0 + rr) * K + k0 + cc * 4);
    *reinterpret_cast<uint2*>(&As[rr][cc * 4]) = d;
  }
#pragma unroll
  for (int e = tid; e < 1024; e += 256) {          // 32 rows x 32 float4
    int rr = e >> 5, cc = e & 31;
    float4 d = *reinterpret_cast<const float4*>(W + (size_t)(k0 + rr) * N + col0 + cc * 4);
    union { _Float16 h[4]; uint2 u; } pk;
    pk.h[0] = (_Float16)d.x; pk.h[1] = (_Float16)d.y;
    pk.h[2] = (_Float16)d.z; pk.h[3] = (_Float16)d.w;
    *reinterpret_cast<uint2*>(&Ws[rr][cc * 4]) = pk.u;
  }
}

// ---------------- Generic WMMA GEMM ----------------
// C[M x N] = A(f16)[M x K] * W(f32->f16)[K x N] + bias (+res) (gelu?)
// block: 256 threads (8 waves); tile 128x128; wave -> 32x64 (8 WMMA accums)
// double-buffered LDS: next K-tile global loads issued before current WMMAs.
template <bool HAS_RES, bool DO_GELU, bool OUT32>
__global__ __launch_bounds__(256)
void gemm_kernel(const _Float16* __restrict__ A, const float* __restrict__ W,
                 const float* __restrict__ bias, const float* __restrict__ res,
                 float* __restrict__ outF, _Float16* __restrict__ outH,
                 int M, int N, int K) {
  __shared__ _Float16 As[2][128][40];    // rows padded to 40 halves (80B, 8B-aligned)
  __shared__ _Float16 Ws[2][32][136];    // rows padded to 136 halves (272B)
  const int tid = threadIdx.x;
  const int lane = tid & 31, w = tid >> 5;
  const int wr = w >> 1, wc = w & 1;                // wave -> 32-row band, 64-col band
  const int row0 = blockIdx.y * 128, col0 = blockIdx.x * 128;

  v8f acc[2][4];
#pragma unroll
  for (int s = 0; s < 2; ++s)
#pragma unroll
    for (int j = 0; j < 4; ++j) acc[s][j] = v8f{};

  const int ksteps = K >> 5;
  stage_tiles(A, W, K, N, row0, col0, 0, tid, As[0], Ws[0]);
  __syncthreads();

  for (int kt = 0; kt < ksteps; ++kt) {
    const int cur = kt & 1;
    if (kt + 1 < ksteps) {
      stage_tiles(A, W, K, N, row0, col0, (kt + 1) << 5, tid, As[cur ^ 1], Ws[cur ^ 1]);
      if (kt + 2 < ksteps) {   // hint the K-tile after next (global_prefetch_b8)
        int rr = tid >> 6, cc = tid & 63;
        __builtin_prefetch(&W[(size_t)(((kt + 2) << 5) + rr) * N + col0 + cc], 0, 1);
      }
    }
    v16h a0 = frag_a(&As[cur][wr * 32][0],      40, lane);
    v16h a1 = frag_a(&As[cur][wr * 32 + 16][0], 40, lane);
#pragma unroll
    for (int j = 0; j < 4; ++j) {
      v16h bj = frag_b(&Ws[cur][0][wc * 64 + j * 16], 136, lane);
      acc[0][j] = wmma16(a0, bj, acc[0][j]);
      acc[1][j] = wmma16(a1, bj, acc[1][j]);
    }
    __syncthreads();
  }

  const int g = lane >> 4, n = lane & 15;
#pragma unroll
  for (int s = 0; s < 2; ++s) {
#pragma unroll
    for (int j = 0; j < 4; ++j) {
      const int col = col0 + wc * 64 + j * 16 + n;
#pragma unroll
      for (int i = 0; i < 8; ++i) {
        const int row = row0 + wr * 32 + s * 16 + i + 8 * g;
        float vv = acc[s][j][i] + bias[col];
        if (HAS_RES) vv += res[(size_t)row * N + col];
        if (DO_GELU) vv = 0.5f * vv * (1.f + erff(vv * 0.70710678118f));
        if (OUT32) outF[(size_t)row * N + col] = vv;
        else       outH[(size_t)row * N + col] = (_Float16)vv;
      }
    }
  }
}

// ---------------- Attention (flash-style, 1 wave / 16 query rows) ----------------
// scores = (Q . V^T) * scale ; probs = softmax ; ctx = probs . V   (reference uses V as keys)
__global__ __launch_bounds__(32)
void attn_kernel(const _Float16* __restrict__ q, const _Float16* __restrict__ v,
                 _Float16* __restrict__ ctx) {
  __shared__ _Float16 Vs[32][64];   // 32 keys x 64 head-dim
  __shared__ _Float16 Ps[16][32];   // probs tile staged for A-fragment transpose
  const int lane = threadIdx.x;
  const int l0 = blockIdx.x * 16;
  const int b = blockIdx.y / NHEAD, h = blockIdx.y % NHEAD;
  const int m = lane & 15, g = lane >> 4;

  // Q fragments for this row block (d=0..31 and d=32..63), kept in registers
  v16h aq0, aq1;
  const _Float16* qbase = q + (size_t)(b * SEQ + l0 + m) * DIM + h * HD;
#pragma unroll
  for (int i = 0; i < 8; ++i) {
    int k0 = frag_k0(i, g);
    aq0[2 * i]     = qbase[k0];
    aq0[2 * i + 1] = qbase[k0 + 1];
    aq1[2 * i]     = qbase[32 + k0];
    aq1[2 * i + 1] = qbase[32 + k0 + 1];
  }

  float rm[8], rl[8];
#pragma unroll
  for (int i = 0; i < 8; ++i) { rm[i] = -__builtin_inff(); rl[i] = 0.f; }
  v8f c0 = {}, c1 = {}, c2 = {}, c3 = {};

  for (int t0 = 0; t0 < SEQ; t0 += 32) {
    // cooperative V-chunk load: lane -> one key row (128B via 8x uint4)
    const uint4* src = reinterpret_cast<const uint4*>(
        v + (size_t)(b * SEQ + t0 + lane) * DIM + h * HD);
    uint4* dst = reinterpret_cast<uint4*>(&Vs[lane][0]);
#pragma unroll
    for (int i = 0; i < 8; ++i) dst[i] = src[i];
    __syncthreads();

    // S = Q(16x64) . Vchunk^T(64x32) -> two 16x16 tiles, 2 K-slices each
    v8f s0 = {}, s1 = {};
    s0 = wmma16(aq0, frag_bt(&Vs[0][0],   64, lane), s0);
    s0 = wmma16(aq1, frag_bt(&Vs[0][32],  64, lane), s0);
    s1 = wmma16(aq0, frag_bt(&Vs[16][0],  64, lane), s1);
    s1 = wmma16(aq1, frag_bt(&Vs[16][32], 64, lane), s1);

    // online softmax over the 32 new columns (reductions within 16-lane groups)
#pragma unroll
    for (int i = 0; i < 8; ++i) {
      float s0i = s0[i] * ATT_SCALE, s1i = s1[i] * ATT_SCALE;
      float mx = fmaxf(s0i, s1i);
#pragma unroll
      for (int msk = 8; msk; msk >>= 1) mx = fmaxf(mx, __shfl_xor(mx, msk));
      float mnew = fmaxf(rm[i], mx);
      float corr = __expf(rm[i] - mnew);
      float p0 = __expf(s0i - mnew), p1 = __expf(s1i - mnew);
      float ps = p0 + p1;
#pragma unroll
      for (int msk = 8; msk; msk >>= 1) ps += __shfl_xor(ps, msk);
      rl[i] = rl[i] * corr + ps;
      rm[i] = mnew;
      c0[i] *= corr; c1[i] *= corr; c2[i] *= corr; c3[i] *= corr;
      Ps[i + 8 * g][m]      = (_Float16)p0;   // transpose C-layout -> A-layout via LDS
      Ps[i + 8 * g][m + 16] = (_Float16)p1;
    }
    __syncthreads();

    // ctx += P(16x32) . Vchunk(32x64)
    v16h ap = frag_a(&Ps[0][0], 32, lane);
    c0 = wmma16(ap, frag_b(&Vs[0][0],  64, lane), c0);
    c1 = wmma16(ap, frag_b(&Vs[0][16], 64, lane), c1);
    c2 = wmma16(ap, frag_b(&Vs[0][32], 64, lane), c2);
    c3 = wmma16(ap, frag_b(&Vs[0][48], 64, lane), c3);
    __syncthreads();
  }

#pragma unroll
  for (int i = 0; i < 8; ++i) {
    const float inv = 1.f / rl[i];
    const int row = l0 + i + 8 * g;
    _Float16* d = ctx + (size_t)(b * SEQ + row) * DIM + h * HD + m;
    d[0]  = (_Float16)(c0[i] * inv);
    d[16] = (_Float16)(c1[i] * inv);
    d[32] = (_Float16)(c2[i] * inv);
    d[48] = (_Float16)(c3[i] * inv);
  }
}

// ---------------- host-side orchestration ----------------
extern "C" void kernel_launch(void* const* d_in, const int* in_sizes, int n_in,
                              void* d_out, int out_size, void* d_ws, size_t ws_size,
                              hipStream_t stream) {
  (void)in_sizes; (void)n_in; (void)out_size; (void)ws_size;
  const float* x    = (const float*)d_in[0];
  const float* ln1g = (const float*)d_in[1];
  const float* ln1b = (const float*)d_in[2];
  const float* Wq   = (const float*)d_in[3];
  const float* bq   = (const float*)d_in[4];
  const float* Wv   = (const float*)d_in[5];
  const float* bv   = (const float*)d_in[6];
  const float* Wo   = (const float*)d_in[7];
  const float* bo   = (const float*)d_in[8];
  const float* ln2g = (const float*)d_in[9];
  const float* ln2b = (const float*)d_in[10];
  const float* W1   = (const float*)d_in[11];
  const float* b1   = (const float*)d_in[12];
  const float* W2   = (const float*)d_in[13];
  const float* b2   = (const float*)d_in[14];

  // workspace layout (~94.4 MB total)
  char* ws = (char*)d_ws;
  size_t off = 0;
  auto alloc = [&](size_t bytes) -> void* {
    void* p = ws + off; off += (bytes + 255) & ~(size_t)255; return p;
  };
  _Float16* h16  = (_Float16*)alloc((size_t)MTOK * DIM * 2);
  _Float16* q16  = (_Float16*)alloc((size_t)MTOK * DIM * 2);
  _Float16* v16p = (_Float16*)alloc((size_t)MTOK * DIM * 2);
  _Float16* c16  = (_Float16*)alloc((size_t)MTOK * DIM * 2);
  _Float16* h216 = (_Float16*)alloc((size_t)MTOK * DIM * 2);
  _Float16* a16  = (_Float16*)alloc((size_t)MTOK * FFDIM * 2);
  float*    skip = (float*)alloc((size_t)MTOK * DIM * 4);
  float*    xA   = (float*)alloc((size_t)MTOK * DIM * 4);
  float*    xB   = (float*)alloc((size_t)MTOK * DIM * 4);

  const dim3 gDD(DIM / 128, MTOK / 128);     // (6, 32)
  const dim3 gF1(FFDIM / 128, MTOK / 128);   // (24, 32)
  const dim3 gAtt(SEQ / 16, BATCH * NHEAD);

  const float* cur = x;
  for (int l = 0; l < NLAYER; ++l) {
    float* nxt = (l == NLAYER - 1) ? (float*)d_out : ((l & 1) ? xB : xA);
    const float* wq = Wq + (size_t)l * DIM * DIM;
    const float* wv = Wv + (size_t)l * DIM * DIM;
    const float* wo = Wo + (size_t)l * DIM * DIM;
    const float* w1 = W1 + (size_t)l * DIM * FFDIM;
    const float* w2 = W2 + (size_t)l * FFDIM * DIM;

    ln_kernel<<<MTOK, 256, 0, stream>>>(cur, ln1g + l * DIM, ln1b + l * DIM, h16);
    gemm_kernel<false, false, false><<<gDD, 256, 0, stream>>>(
        h16, wq, bq + l * DIM, nullptr, nullptr, q16, MTOK, DIM, DIM);
    gemm_kernel<false, false, false><<<gDD, 256, 0, stream>>>(
        h16, wv, bv + l * DIM, nullptr, nullptr, v16p, MTOK, DIM, DIM);
    attn_kernel<<<gAtt, 32, 0, stream>>>(q16, v16p, c16);
    gemm_kernel<true, false, true><<<gDD, 256, 0, stream>>>(
        c16, wo, bo + l * DIM, cur, skip, nullptr, MTOK, DIM, DIM);
    ln_kernel<<<MTOK, 256, 0, stream>>>(skip, ln2g + l * DIM, ln2b + l * DIM, h216);
    gemm_kernel<false, true, false><<<gF1, 256, 0, stream>>>(
        h216, w1, b1 + l * FFDIM, nullptr, nullptr, a16, MTOK, FFDIM, DIM);
    gemm_kernel<true, false, true><<<gDD, 256, 0, stream>>>(
        a16, w2, b2 + l * DIM, skip, nxt, nullptr, MTOK, DIM, FFDIM);
    cur = nxt;
  }
}
